// LSTM_32530082299983
// MI455X (gfx1250) — compile-verified, matrix operand order checked by reading, not run on hardware
//
#include <hip/hip_runtime.h>
#include <hip/hip_bf16.h>
#include <cstddef>
#include <cstdint>

// ---------------------------------------------------------------------------
// Types for CDNA5 WMMA (wave32): v_wmma_f32_16x16x32_bf16
// ---------------------------------------------------------------------------
typedef __bf16 bf16_t;
typedef __attribute__((ext_vector_type(16))) __bf16 v16bf;
typedef __attribute__((ext_vector_type(8)))  __bf16 v8bf;
typedef __attribute__((ext_vector_type(8)))  float  v8f;

#define BATCH  512
#define TSTEPS 256
#define HDIM   256
#define GDIM   1024   // 4*H
#define BROWS  32     // batch rows per LSTM workgroup

__device__ __forceinline__ float gelu_f(float x) {
  return 0.5f * x * (1.0f + erff(x * 0.70710678118654752f));
}
__device__ __forceinline__ float sigmoid_f(float x) {
  return 1.0f / (1.0f + __expf(-x));
}

// Load a 16-element bf16 fragment slice for one lane from row-major memory.
// Fragment element e maps to column base0 + (e<8 ? e : e+8); i.e. two
// contiguous 16-byte chunks -> two b128 global loads.
__device__ __forceinline__ v16bf load_frag16(const bf16_t* p) {
  v8bf lo = *(const v8bf*)(p);
  v8bf hi = *(const v8bf*)(p + 16);
  v16bf r;
#pragma unroll
  for (int i = 0; i < 8; ++i) { r[i] = lo[i]; r[i + 8] = hi[i]; }
  return r;
}

__device__ __forceinline__ v8f wmma_bf16(v16bf a, v16bf b, v8f c) {
  // (neg_a, A, neg_b, B, c_mod, C, reuse_a, reuse_b)
  return __builtin_amdgcn_wmma_f32_16x16x32_bf16(false, a, false, b, (short)0, c,
                                                 false, false);
}

// ---------------------------------------------------------------------------
// CDNA5 async global->LDS copy (VGLOBAL GLOBAL_LOAD_ASYNC_TO_LDS_B128,
// tracked by ASYNCcnt). VDST = per-lane LDS byte address, VADDR = 64-bit
// global address ("GV" mode). Generic pointers to __shared__ carry the LDS
// offset in their low 32 bits.
// ---------------------------------------------------------------------------
__device__ __forceinline__ void async_ld_b128(void* lds, const void* gaddr) {
  unsigned loff = (unsigned)(uintptr_t)lds;
  asm volatile("global_load_async_to_lds_b128 %0, %1, off"
               :
               : "v"(loff), "v"(gaddr)
               : "memory");
}
__device__ __forceinline__ void wait_async0() {
#if __has_builtin(__builtin_amdgcn_s_wait_asynccnt)
  __builtin_amdgcn_s_wait_asynccnt(0);
#else
  asm volatile("s_wait_asynccnt 0x0" ::: "memory");
#endif
}

// ---------------------------------------------------------------------------
// Kernel 1: weight prep — convert LSTM + decoder weights to bf16, sum biases
// ---------------------------------------------------------------------------
__global__ __launch_bounds__(256) void prep_kernel(
    const float* __restrict__ Wih, const float* __restrict__ Whh,
    const float* __restrict__ bih, const float* __restrict__ bhh,
    const float* __restrict__ dw1,
    bf16_t* __restrict__ WihB, bf16_t* __restrict__ WhhB,
    bf16_t* __restrict__ dw1B, float* __restrict__ biasSum) {
  int i = blockIdx.x * 256 + threadIdx.x;
  const int NW = 3 * GDIM * HDIM;
  if (i < NW) {
    WihB[i] = (bf16_t)Wih[i];
    WhhB[i] = (bf16_t)Whh[i];
  }
  if (i < 128 * HDIM) dw1B[i] = (bf16_t)dw1[i];
  if (i < 3 * GDIM)   biasSum[i] = bih[i] + bhh[i];
}

// ---------------------------------------------------------------------------
// Kernel 2: encoder — 2 tiny GEMMs + GELU + LayerNorm, one block per row
// ---------------------------------------------------------------------------
__global__ __launch_bounds__(256) void encoder_kernel(
    const float* __restrict__ x,
    const float* __restrict__ w1, const float* __restrict__ b1,
    const float* __restrict__ g1, const float* __restrict__ be1,
    const float* __restrict__ w2, const float* __restrict__ b2,
    const float* __restrict__ g2, const float* __restrict__ be2,
    bf16_t* __restrict__ h2out) {
  __shared__ float xs[9];
  __shared__ float h1s[128];
  __shared__ float red[256];
  const int b = blockIdx.x, tid = threadIdx.x;
  if (tid < 9) xs[tid] = x[b * 9 + tid];
  __syncthreads();

  float v1 = 0.f;
  if (tid < 128) {
    float a = b1[tid];
#pragma unroll
    for (int i = 0; i < 9; ++i) a += xs[i] * w1[tid * 9 + i];
    v1 = gelu_f(a);
  }
  red[tid] = (tid < 128) ? v1 : 0.f;
  __syncthreads();
  for (int s = 128; s > 0; s >>= 1) { if (tid < s) red[tid] += red[tid + s]; __syncthreads(); }
  float mu = red[0] * (1.f / 128.f);
  __syncthreads();
  float d1 = (tid < 128) ? (v1 - mu) : 0.f;
  red[tid] = d1 * d1;
  __syncthreads();
  for (int s = 128; s > 0; s >>= 1) { if (tid < s) red[tid] += red[tid + s]; __syncthreads(); }
  float rs = rsqrtf(red[0] * (1.f / 128.f) + 1e-5f);
  __syncthreads();
  if (tid < 128) h1s[tid] = d1 * rs * g1[tid] + be1[tid];
  __syncthreads();

  float a2 = b2[tid];
  for (int i = 0; i < 128; ++i) a2 += h1s[i] * w2[tid * 128 + i];
  float v2 = gelu_f(a2);
  red[tid] = v2;
  __syncthreads();
  for (int s = 128; s > 0; s >>= 1) { if (tid < s) red[tid] += red[tid + s]; __syncthreads(); }
  mu = red[0] * (1.f / 256.f);
  __syncthreads();
  float d2 = v2 - mu;
  red[tid] = d2 * d2;
  __syncthreads();
  for (int s = 128; s > 0; s >>= 1) { if (tid < s) red[tid] += red[tid + s]; __syncthreads(); }
  rs = rsqrtf(red[0] * (1.f / 256.f) + 1e-5f);
  float hn = d2 * rs * g2[tid] + be2[tid];
  h2out[b * HDIM + tid] = (bf16_t)hn;
}

// ---------------------------------------------------------------------------
// Kernel 3: xg0 = h2 @ Wih0^T + (bih0+bhh0)   (layer-0 input gates, time-
// invariant because the LSTM input sequence is a broadcast of h2).
// ---------------------------------------------------------------------------
__global__ __launch_bounds__(256) void xgate0_kernel(
    const bf16_t* __restrict__ h2b, const bf16_t* __restrict__ Wih0b,
    const float* __restrict__ bias0, float* __restrict__ xg0) {
  const int tid = threadIdx.x;
  const int lane = tid & 31;
  const int wid = (blockIdx.x * 256 + tid) >> 5;  // 0..2047
  const int mt = wid >> 6;                        // 0..31 row tiles
  const int ct = wid & 63;                        // 0..63 col tiles
  const int nlo = lane & 15, khalf = lane >> 4;
  v8f acc;
#pragma unroll
  for (int r = 0; r < 8; ++r) acc[r] = 0.f;
#pragma unroll
  for (int kt = 0; kt < 8; ++kt) {
    const int base0 = kt * 32 + khalf * 8;
    v16bf a  = load_frag16(h2b + (size_t)(mt * 16 + nlo) * HDIM + base0);
    v16bf bm = load_frag16(Wih0b + (size_t)(ct * 16 + nlo) * HDIM + base0);
    acc = wmma_bf16(a, bm, acc);
  }
  const float bv = bias0[ct * 16 + nlo];
#pragma unroll
  for (int r = 0; r < 8; ++r) {
    const int mrow = mt * 16 + r + 8 * khalf;     // C layout: M = r + 8*(lane>=16)
    xg0[(size_t)mrow * GDIM + ct * 16 + nlo] = acc[r] + bv;
  }
}

// ---------------------------------------------------------------------------
// Kernel 4: persistent LSTM layer. 16 WGs x 8 waves, 32 batch rows per WG.
// h_t lives in LDS in A-fragment layout; c_t in accumulator VGPRs; weights
// stream from L2 as B-fragments. x_t (layers 1,2) is double-buffered in LDS
// via GLOBAL_LOAD_ASYNC_TO_LDS_B128: step t+1's input is fetched by the TDM-
// style async path while step t's WMMAs run, and retired with
// s_wait_asynccnt + barrier at end of step.
// ---------------------------------------------------------------------------
template <bool FIRST>
__global__ __launch_bounds__(256) void lstm_kernel(
    const float* __restrict__ xg0,      // FIRST: [B,4H] precomputed gates
    const bf16_t* xseq,                 // !FIRST: [B,T,H] bf16 inputs (in-place)
    const bf16_t* __restrict__ Whh,     // [4H,H] bf16
    const bf16_t* __restrict__ Wih,     // !FIRST: [4H,H] bf16
    const float* __restrict__ bias,     // !FIRST: bih+bhh [4H]
    bf16_t* outseq)                     // [B,T,H] bf16
{
  __shared__ v16bf hA[2][8][32];        // h_t fragments: [rowtile][ktile][lane]
  __shared__ v16bf xA[2][2][8][32];     // x_t fragments: [buf][rowtile][ktile][lane]

  const int tid = threadIdx.x;
  const int lane = tid & 31;
  const int wv = tid >> 5;            // wave 0..7
  const int bw = blockIdx.x * BROWS;  // batch row base
  const int khalf = lane >> 4;
  const int nlo = lane & 15;

  // zero h_0 fragments
  {
    unsigned* hz = (unsigned*)hA;
    for (int i = tid; i < 2 * 8 * 32 * 8; i += 256) hz[i] = 0u;
  }

  // gate-column tile assignment for this wave
  int colArr[8];
#pragma unroll
  for (int s = 0; s < 8; ++s) {
    const int ht = s >> 2, g = s & 3;
    const int ctile = (wv + 8 * ht) + 16 * g;
    colArr[s] = ctile * 16 + nlo;
  }
  float bs[8];
  if (!FIRST) {
#pragma unroll
    for (int s = 0; s < 8; ++s) bs[s] = bias[colArr[s]];
  }

  // per-thread async-copy assignment (2 fragment-lanes per thread):
  // frag-lane idx -> (rowtile, ktile, dest lane); global source is two
  // contiguous 16B chunks of row m at column kt*32 + kh*8 (+16).
  const bf16_t* gsrc[2];
  void*         ldst[2];
  if (!FIRST) {
#pragma unroll
    for (int u = 0; u < 2; ++u) {
      const int idx = tid + u * 256;
      const int rt = idx >> 8;
      const int kt = (idx >> 5) & 7;
      const int l  = idx & 31;
      const int m  = bw + rt * 16 + (l & 15);
      const int kh = l >> 4;
      gsrc[u] = xseq + (size_t)m * (TSTEPS * HDIM) + kt * 32 + kh * 8;
      ldst[u] = (void*)&xA[0][rt][kt][l];
    }
  }

  v8f cst[2][2];  // cell state [rowtile][hiddentile]
#pragma unroll
  for (int rt = 0; rt < 2; ++rt)
#pragma unroll
    for (int ht = 0; ht < 2; ++ht)
#pragma unroll
      for (int r = 0; r < 8; ++r) cst[rt][ht][r] = 0.f;

  // prologue: async-fetch x_0 into buffer 0
  if (!FIRST) {
#pragma unroll
    for (int u = 0; u < 2; ++u) {
      async_ld_b128(ldst[u], gsrc[u]);
      async_ld_b128((char*)ldst[u] + 16, gsrc[u] + 16);
    }
    wait_async0();
  }
  __syncthreads();

#pragma unroll 1
  for (int t = 0; t < TSTEPS; ++t) {
    const int cur = t & 1;
    if (!FIRST && (t + 1) < TSTEPS) {
      // overlap: async-fetch x_{t+1} into the other buffer while we compute
      const size_t goff = (size_t)(t + 1) * HDIM;
      const size_t boff = (size_t)(cur ^ 1) * (2 * 8 * 32 * sizeof(v16bf));
#pragma unroll
      for (int u = 0; u < 2; ++u) {
        const bf16_t* g = gsrc[u] + goff;
        char* l = (char*)ldst[u] + boff;
        async_ld_b128(l, g);
        async_ld_b128(l + 16, g + 16);
      }
    }

    // gates = init + h_{t-1}@Whh^T (+ x_t@Wih^T)
    v8f acc[2][8];
    if (FIRST) {
#pragma unroll
      for (int rt = 0; rt < 2; ++rt)
#pragma unroll
        for (int s = 0; s < 8; ++s)
#pragma unroll
          for (int r = 0; r < 8; ++r) {
            const int mrow = bw + rt * 16 + r + 8 * khalf;
            acc[rt][s][r] = xg0[(size_t)mrow * GDIM + colArr[s]];  // L2-resident
          }
    } else {
#pragma unroll
      for (int rt = 0; rt < 2; ++rt)
#pragma unroll
        for (int s = 0; s < 8; ++s)
#pragma unroll
          for (int r = 0; r < 8; ++r) acc[rt][s][r] = bs[s];
    }

#pragma unroll
    for (int kt = 0; kt < 8; ++kt) {
      const int base0 = kt * 32 + khalf * 8;
      v16bf aH0 = hA[0][kt][lane];
      v16bf aH1 = hA[1][kt][lane];
      v16bf aX0, aX1;
      if (!FIRST) { aX0 = xA[cur][0][kt][lane]; aX1 = xA[cur][1][kt][lane]; }
#pragma unroll
      for (int s = 0; s < 8; ++s) {
        const int nrow = colArr[s];  // = ct*16 + nlo
        v16bf bH = load_frag16(Whh + (size_t)nrow * HDIM + base0);
        acc[0][s] = wmma_bf16(aH0, bH, acc[0][s]);
        acc[1][s] = wmma_bf16(aH1, bH, acc[1][s]);
        if (!FIRST) {
          v16bf bX = load_frag16(Wih + (size_t)nrow * HDIM + base0);
          acc[0][s] = wmma_bf16(aX0, bX, acc[0][s]);
          acc[1][s] = wmma_bf16(aX1, bX, acc[1][s]);
        }
      }
    }
    __syncthreads();  // all hA/xA[cur] reads for step t done before overwrite

    // elementwise gate math; write h_t to LDS fragments + global sequence
#pragma unroll
    for (int rt = 0; rt < 2; ++rt) {
#pragma unroll
      for (int ht = 0; ht < 2; ++ht) {
        v8f gi = acc[rt][ht * 4 + 0];
        v8f gf = acc[rt][ht * 4 + 1];
        v8f gg = acc[rt][ht * 4 + 2];
        v8f go = acc[rt][ht * 4 + 3];
        const int ncol = (wv + 8 * ht) * 16 + nlo;
        const int kt2 = ncol >> 5;
        const int kk = ncol & 31;
        const int e = (kk & 7) + ((kk & 16) ? 8 : 0);
        const int lhalf = (kk >> 3) & 1;
#pragma unroll
        for (int r = 0; r < 8; ++r) {
          const float iv = sigmoid_f(gi[r]);
          const float fv = sigmoid_f(gf[r]);
          const float gv = tanhf(gg[r]);
          const float ov = sigmoid_f(go[r]);
          const float cn = fv * cst[rt][ht][r] + iv * gv;
          cst[rt][ht][r] = cn;
          const float hv = ov * tanhf(cn);
          const int mloc = r + 8 * khalf;  // 0..15 in row tile
          outseq[(size_t)(bw + rt * 16 + mloc) * (TSTEPS * HDIM) +
                 (size_t)t * HDIM + ncol] = (bf16_t)hv;
          const int lt = mloc + (lhalf << 4);
          ((bf16_t*)&hA[rt][kt2][lt])[e] = (bf16_t)hv;  // scatter into A-frag
        }
      }
    }
    if (!FIRST) wait_async0();  // this wave's async LDS writes have landed
    __syncthreads();            // hA + xA[next] visible to all waves
  }
}

// ---------------------------------------------------------------------------
// Kernel 5: decoder. WMMA GEMM [BT,256]x[256,128] + GELU, then the tiny
// 128->2 projection via per-lane dot products from LDS.
// ---------------------------------------------------------------------------
__global__ __launch_bounds__(256) void decoder_kernel(
    const bf16_t* __restrict__ h3, const bf16_t* __restrict__ w1b,
    const float* __restrict__ b1, const float* __restrict__ w2,
    const float* __restrict__ b2, float* __restrict__ out) {
  __shared__ float dbuf[8][16][130];
  const int tid = threadIdx.x, lane = tid & 31, wv = tid >> 5;
  const int nlo = lane & 15, khalf = lane >> 4;
  const size_t rowbase = (size_t)blockIdx.x * 128 + (size_t)wv * 16;

  v8f acc[8];
#pragma unroll
  for (int ctile = 0; ctile < 8; ++ctile) {
    const float bv = b1[ctile * 16 + nlo];
#pragma unroll
    for (int r = 0; r < 8; ++r) acc[ctile][r] = bv;
  }
#pragma unroll
  for (int kt = 0; kt < 8; ++kt) {
    const int base0 = kt * 32 + khalf * 8;
    v16bf a = load_frag16(h3 + (rowbase + nlo) * HDIM + base0);
#pragma unroll
    for (int ctile = 0; ctile < 8; ++ctile) {
      v16bf bm = load_frag16(w1b + (size_t)(ctile * 16 + nlo) * HDIM + base0);
      acc[ctile] = wmma_bf16(a, bm, acc[ctile]);
    }
  }
#pragma unroll
  for (int ctile = 0; ctile < 8; ++ctile)
#pragma unroll
    for (int r = 0; r < 8; ++r)
      dbuf[wv][r + 8 * khalf][ctile * 16 + nlo] = gelu_f(acc[ctile][r]);
  __syncthreads();

  const int row = lane >> 1, j = lane & 1;
  float s = b2[j];
  for (int k = 0; k < 128; ++k) s += dbuf[wv][row][k] * w2[j * 128 + k];
  out[(rowbase + row) * 2 + j] = s;
}

// ---------------------------------------------------------------------------
// Host launcher
// ---------------------------------------------------------------------------
extern "C" void kernel_launch(void* const* d_in, const int* in_sizes, int n_in,
                              void* d_out, int out_size, void* d_ws, size_t ws_size,
                              hipStream_t stream) {
  const float* x      = (const float*)d_in[0];
  const float* enc_w1 = (const float*)d_in[1];
  const float* enc_b1 = (const float*)d_in[2];
  const float* ln1_g  = (const float*)d_in[3];
  const float* ln1_b  = (const float*)d_in[4];
  const float* enc_w2 = (const float*)d_in[5];
  const float* enc_b2 = (const float*)d_in[6];
  const float* ln2_g  = (const float*)d_in[7];
  const float* ln2_b  = (const float*)d_in[8];
  const float* Wih    = (const float*)d_in[9];
  const float* Whh    = (const float*)d_in[10];
  const float* bih    = (const float*)d_in[11];
  const float* bhh    = (const float*)d_in[12];
  const float* dec_w1 = (const float*)d_in[13];
  const float* dec_b1 = (const float*)d_in[14];
  const float* dec_w2 = (const float*)d_in[15];
  const float* dec_b2 = (const float*)d_in[16];
  float* out = (float*)d_out;

  char* ws = (char*)d_ws;
  size_t off = 0;
  auto alloc = [&](size_t bytes) -> void* {
    void* p = ws + off;
    off = (off + bytes + 255) & ~(size_t)255;
    return p;
  };
  bf16_t* WihB    = (bf16_t*)alloc((size_t)3 * GDIM * HDIM * sizeof(bf16_t));
  bf16_t* WhhB    = (bf16_t*)alloc((size_t)3 * GDIM * HDIM * sizeof(bf16_t));
  bf16_t* dw1B    = (bf16_t*)alloc((size_t)128 * HDIM * sizeof(bf16_t));
  float*  biasSum = (float*)alloc((size_t)3 * GDIM * sizeof(float));
  bf16_t* h2B     = (bf16_t*)alloc((size_t)BATCH * HDIM * sizeof(bf16_t));
  float*  xg0     = (float*)alloc((size_t)BATCH * GDIM * sizeof(float));
  bf16_t* seq     = (bf16_t*)alloc((size_t)BATCH * TSTEPS * HDIM * sizeof(bf16_t));

  prep_kernel<<<(3 * GDIM * HDIM + 255) / 256, 256, 0, stream>>>(
      Wih, Whh, bih, bhh, dec_w1, WihB, WhhB, dw1B, biasSum);
  encoder_kernel<<<BATCH, 256, 0, stream>>>(
      x, enc_w1, enc_b1, ln1_g, ln1_b, enc_w2, enc_b2, ln2_g, ln2_b, h2B);
  xgate0_kernel<<<256, 256, 0, stream>>>(h2B, WihB, biasSum, xg0);

  lstm_kernel<true><<<BATCH / BROWS, 256, 0, stream>>>(
      xg0, (const bf16_t*)nullptr, WhhB, (const bf16_t*)nullptr,
      (const float*)nullptr, seq);
  lstm_kernel<false><<<BATCH / BROWS, 256, 0, stream>>>(
      (const float*)nullptr, seq, WhhB + (size_t)GDIM * HDIM,
      WihB + (size_t)GDIM * HDIM, biasSum + GDIM, seq);
  lstm_kernel<false><<<BATCH / BROWS, 256, 0, stream>>>(
      (const float*)nullptr, seq, WhhB + (size_t)2 * GDIM * HDIM,
      WihB + (size_t)2 * GDIM * HDIM, biasSum + 2 * GDIM, seq);

  decoder_kernel<<<(BATCH * TSTEPS) / 128, 256, 0, stream>>>(
      seq, dw1B, dec_b1, dec_w2, dec_b2, out);
}
